// MaskedAutoencoderViT_33337536152114
// MI455X (gfx1250) — matrix operand order repaired and checked by reading, not compiled, not run on hardware
//
#include <hip/hip_runtime.h>

// ---------------------------------------------------------------------------
// CDNA5 (gfx1250, wave32) transformer layer.
// - Big GEMMs: v_wmma_f32_16x16x32_f16, fp32 accumulate, 128x128x64 tiles.
//   Global->LDS staging uses GLOBAL_LOAD_ASYNC_TO_LDS_B128 (ASYNCcnt) with
//   double-buffered fp32 tiles (128KB LDS); f32->f16 conversion happens at
//   fragment load via ds_load_b128 + v_cvt_pk_f16_f32 (co-executes with XDL).
//   All four B fragments kept live per k-chunk to avoid WAR hazard NOPs and
//   per-load dscnt stalls.
// - Attention: WMMA for QK^T and PV, 64x512 score panel + 128x512 transposed
//   V panel resident in the 320KB WGP LDS.
// ---------------------------------------------------------------------------

typedef __attribute__((ext_vector_type(16))) _Float16 v16h;
typedef __attribute__((ext_vector_type(8)))  _Float16 v8h;
typedef __attribute__((ext_vector_type(8)))  float    v8f;

#define B_   4
#define S_   512
#define D_   4096
#define H_   32
#define HD_  128
#define P_   10
#define R_   16
#define HID_ 11008
#define EPS_ 1e-5f
#define SCALE_ 0.08838834764831845f   // 1/sqrt(128)

__device__ inline v8f wmma_zero() {
  v8f z;
#pragma unroll
  for (int i = 0; i < 8; ++i) z[i] = 0.0f;
  return z;
}

__device__ inline v8f wmma16(v16h a, v16h b, v8f c) {
  // D = A(16x32 f16) x B(32x16 f16) + C(16x16 f32)
  return __builtin_amdgcn_wmma_f32_16x16x32_f16(
      /*neg_a=*/false, a, /*neg_b=*/false, b,
      /*c_mod=*/(short)0, c, /*reuse_a=*/false, /*reuse_b=*/false);
}

// Async DMA: global memory -> LDS, 16 bytes per lane, tracked by ASYNCcnt.
__device__ inline void async_ld_b128(unsigned lds_off, const void* gaddr) {
  asm volatile("global_load_async_to_lds_b128 %0, %1, off"
               :: "v"(lds_off), "v"(gaddr) : "memory");
}
__device__ inline void wait_async_le16() {
  asm volatile("s_wait_asynccnt 16" ::: "memory");
}
__device__ inline void wait_async_0() {
  asm volatile("s_wait_asynccnt 0" ::: "memory");
}

// ---- f16-in-LDS fragment loaders (used by attention) ----------------------
// A fragment (16x32 f16, row-major, leading dim ld):
// lanes 0-15: M=lane,    K = k0 + 0..15
// lanes 16-31: M=lane-16, K = k0 + 16..31
__device__ inline v16h lds_a_frag(const _Float16* base, int ld, int k0) {
  int lane = threadIdx.x & 31;
  const _Float16* p = base + (lane & 15) * ld + k0 + ((lane >> 4) << 4);
  v8h lo = *(const v8h*)p;
  v8h hi = *(const v8h*)(p + 8);
  v16h r;
#pragma unroll
  for (int i = 0; i < 8; ++i) { r[i] = lo[i]; r[i + 8] = hi[i]; }
  return r;
}
// B fragment (32x16 f16, column n K-contiguous, leading dim ld):
// lanes 0-15: N=lane, K=k0+0..15 ; lanes 16-31: N=lane-16, K=k0+16..31
__device__ inline v16h lds_b_frag(const _Float16* base, int ld, int k0) {
  int lane = threadIdx.x & 31;
  const _Float16* p = base + (lane & 15) * ld + k0 + ((lane >> 4) << 4);
  v8h lo = *(const v8h*)p;
  v8h hi = *(const v8h*)(p + 8);
  v16h r;
#pragma unroll
  for (int i = 0; i < 8; ++i) { r[i] = lo[i]; r[i + 8] = hi[i]; }
  return r;
}

// ---- f32-in-LDS fragment loaders (GEMM path: ds_load_b128 + cvt) ----------
// leading dim is GEMM BK = 64 floats.
__device__ inline v16h lds_a_frag_f32(const float* base, int k0) {
  int lane = threadIdx.x & 31;
  const float* p = base + (lane & 15) * 64 + k0 + ((lane >> 4) << 4);
  float4 f0 = *(const float4*)(p);
  float4 f1 = *(const float4*)(p + 4);
  float4 f2 = *(const float4*)(p + 8);
  float4 f3 = *(const float4*)(p + 12);
  v16h r;
  r[0]  = (_Float16)f0.x; r[1]  = (_Float16)f0.y;
  r[2]  = (_Float16)f0.z; r[3]  = (_Float16)f0.w;
  r[4]  = (_Float16)f1.x; r[5]  = (_Float16)f1.y;
  r[6]  = (_Float16)f1.z; r[7]  = (_Float16)f1.w;
  r[8]  = (_Float16)f2.x; r[9]  = (_Float16)f2.y;
  r[10] = (_Float16)f2.z; r[11] = (_Float16)f2.w;
  r[12] = (_Float16)f3.x; r[13] = (_Float16)f3.y;
  r[14] = (_Float16)f3.z; r[15] = (_Float16)f3.w;
  return r;
}
__device__ inline v16h lds_b_frag_f32(const float* base, int k0) {
  int lane = threadIdx.x & 31;
  const float* p = base + (lane & 15) * 64 + k0 + ((lane >> 4) << 4);
  float4 f0 = *(const float4*)(p);
  float4 f1 = *(const float4*)(p + 4);
  float4 f2 = *(const float4*)(p + 8);
  float4 f3 = *(const float4*)(p + 12);
  v16h r;
  r[0]  = (_Float16)f0.x; r[1]  = (_Float16)f0.y;
  r[2]  = (_Float16)f0.z; r[3]  = (_Float16)f0.w;
  r[4]  = (_Float16)f1.x; r[5]  = (_Float16)f1.y;
  r[6]  = (_Float16)f1.z; r[7]  = (_Float16)f1.w;
  r[8]  = (_Float16)f2.x; r[9]  = (_Float16)f2.y;
  r[10] = (_Float16)f2.z; r[11] = (_Float16)f2.w;
  r[12] = (_Float16)f3.x; r[13] = (_Float16)f3.y;
  r[14] = (_Float16)f3.z; r[15] = (_Float16)f3.w;
  return r;
}

// ---------------------------------------------------------------------------
// C(MxN) = A(MxK) @ W(NxK)^T (+bias). fp32 in/out, f16 WMMA math.
// 256 threads (8 waves), 128x128 tile, BK=64, double-buffered async staging.
// Dynamic LDS: 2 x (32KB A + 32KB W) = 128KB (2 blocks / 320KB WGP).
// Each thread issues exactly 16 async B128 loads per tile; in-order ASYNCcnt
// completion makes "s_wait_asynccnt 16" mean "previous tile fully landed".
// 16 WMMAs per barrier pair; 4 live B fragments for load/WMMA overlap.
// ---------------------------------------------------------------------------
__global__ __launch_bounds__(256)
void gemm_f16(const float* __restrict__ A, const float* __restrict__ W,
              const float* __restrict__ bias, float* __restrict__ C,
              int M, int N, int K) {
  extern __shared__ char gsm[];   // [2][ A:32KB | W:32KB ]
  const int tid  = threadIdx.x;
  const int lane = tid & 31;
  const int wave = tid >> 5;
  const int wm   = wave >> 1;   // 0..3 : 32-row slice
  const int wn   = wave & 1;    // 0..1 : 64-col slice
  const int row0 = blockIdx.y * 128;
  const int col0 = blockIdx.x * 128;

  v8f acc[2][4];
#pragma unroll
  for (int i = 0; i < 2; ++i)
#pragma unroll
    for (int j = 0; j < 4; ++j) acc[i][j] = wmma_zero();

  // staging map per tile: A = 128 rows x 64 floats = 2048 16B-segments,
  // W likewise; 8 A + 8 W segments per thread.
  // seg = tid + i*256 ; row = seg>>4 ; col = (seg&15)*4 floats
  auto issue_tile = [&](int k0, int buf) {
    char* base = gsm + buf * (64 * 1024);
#pragma unroll
    for (int i = 0; i < 8; ++i) {
      int seg = tid + i * 256;
      int r = seg >> 4;
      int c = (seg & 15) << 2;
      int gr = row0 + r;
      if (gr >= M) gr = M - 1;  // clamp: always issue, keeps ASYNCcnt in lockstep
      async_ld_b128((unsigned)(size_t)(base + (r * 64 + c) * 4),
                    A + (size_t)gr * K + k0 + c);
      async_ld_b128((unsigned)(size_t)(base + 32 * 1024 + (r * 64 + c) * 4),
                    W + (size_t)(col0 + r) * K + k0 + c);
    }
  };

  issue_tile(0, 0);
  const int KT = K / 64;
  for (int kt = 0; kt < KT; ++kt) {
    const int cur = kt & 1;
    if (kt + 1 < KT) {
      issue_tile((kt + 1) * 64, cur ^ 1);  // prefetch next tile
      wait_async_le16();                   // oldest 16 (= current tile) landed
    } else {
      wait_async_0();
    }
    __syncthreads();

    const float* As = (const float*)(gsm + cur * 65536) + (wm * 32) * 64;
    const float* Ws = (const float*)(gsm + cur * 65536 + 32768) + (wn * 64) * 64;
#pragma unroll
    for (int kc = 0; kc < 2; ++kc) {
      const int k0 = kc * 32;
      // A fragments (reused by all 4 column tiles)
      v16h a0 = lds_a_frag_f32(As, k0);
      v16h a1 = lds_a_frag_f32(As + 16 * 64, k0);
      // keep all 4 B fragments live: loads overlap WMMA issue, no WAR NOPs
      v16h b0 = lds_b_frag_f32(Ws + 0 * 16 * 64, k0);
      v16h b1 = lds_b_frag_f32(Ws + 1 * 16 * 64, k0);
      acc[0][0] = wmma16(a0, b0, acc[0][0]);
      acc[1][0] = wmma16(a1, b0, acc[1][0]);
      v16h b2 = lds_b_frag_f32(Ws + 2 * 16 * 64, k0);
      acc[0][1] = wmma16(a0, b1, acc[0][1]);
      acc[1][1] = wmma16(a1, b1, acc[1][1]);
      v16h b3 = lds_b_frag_f32(Ws + 3 * 16 * 64, k0);
      acc[0][2] = wmma16(a0, b2, acc[0][2]);
      acc[1][2] = wmma16(a1, b2, acc[1][2]);
      acc[0][3] = wmma16(a0, b3, acc[0][3]);
      acc[1][3] = wmma16(a1, b3, acc[1][3]);
    }
    __syncthreads();   // everyone done with buffer `cur` before it is refilled
  }

  // C/D layout: VGPR r -> M=r (lanes 0-15) / M=8+r (lanes 16-31); N = lane&15
  const int cl = lane & 15;
  const int ch = lane >> 4;
#pragma unroll
  for (int mt = 0; mt < 2; ++mt)
#pragma unroll
    for (int nt = 0; nt < 4; ++nt)
#pragma unroll
      for (int r = 0; r < 8; ++r) {
        int m = row0 + wm * 32 + mt * 16 + ch * 8 + r;
        int n = col0 + wn * 64 + nt * 16 + cl;
        if (m < M) {
          float v = acc[mt][nt][r];
          if (bias) v += bias[n];
          C[(size_t)m * N + n] = v;
        }
      }
}

// ---------------------------------------------------------------------------
// Attention: one block per (64 q rows, head, batch). 128 threads = 4 waves,
// each wave owns 16 q rows. LDS: q(f16 64x128) | scores(f32 64x512, reused
// as Vt f16 128x512) | probs(f16 64x512) | prompt-probs(f32 64x10).
// ---------------------------------------------------------------------------
__global__ __launch_bounds__(128)
void attn_f16(const float* __restrict__ Q, const float* __restrict__ Kc,
              const float* __restrict__ Vc, const float* __restrict__ PK,
              const float* __restrict__ PV, const float* __restrict__ gate,
              const float* __restrict__ new_gate, float* __restrict__ O) {
  extern __shared__ char smem[];
  _Float16* qs = (_Float16*)smem;                                 // 64x128 f16
  float*    sc = (float*)(smem + 64 * 128 * 2);                   // 64x512 f32
  _Float16* vt = (_Float16*)sc;                                   // 128x512 f16 (reuse)
  _Float16* pr = (_Float16*)(smem + 64 * 128 * 2 + 64 * 512 * 4); // 64x512 f16
  float*    ps = (float*)(smem + 64 * 128 * 2 + 64 * 512 * 4 + 64 * 512 * 2);

  const int tid  = threadIdx.x;
  const int lane = tid & 31;
  const int wave = tid >> 5;
  const int q0 = blockIdx.x * 64;
  const int h  = blockIdx.y;
  const int b  = blockIdx.z;
  const int cl = lane & 15;
  const int ch = lane >> 4;

  // stage Q tile as f16
  for (int e = tid * 4; e < 64 * 128; e += 128 * 4) {
    int r = e >> 7, c = e & 127;
    float4 f = *(const float4*)(Q + (size_t)(b * S_ + q0 + r) * D_ + h * HD_ + c);
    qs[r * 128 + c + 0] = (_Float16)f.x; qs[r * 128 + c + 1] = (_Float16)f.y;
    qs[r * 128 + c + 2] = (_Float16)f.z; qs[r * 128 + c + 3] = (_Float16)f.w;
  }
  __syncthreads();

  // ---- scores = Q @ K^T (per-wave 16 rows x all 512 keys) ----
  v16h aq[4];
#pragma unroll
  for (int kc = 0; kc < 4; ++kc)
    aq[kc] = lds_a_frag(qs + (wave * 16) * 128, 128, kc * 32);

  for (int n0 = 0; n0 < S_; n0 += 16) {
    v8f acc = wmma_zero();
    int key = n0 + cl;
#pragma unroll
    for (int kc = 0; kc < 4; ++kc) {
      const float* kp = Kc + (size_t)(b * S_ + key) * D_ + h * HD_ + kc * 32 + ch * 16;
      v16h bf;
#pragma unroll
      for (int i = 0; i < 16; i += 4) {
        float4 f = *(const float4*)(kp + i);
        bf[i] = (_Float16)f.x; bf[i + 1] = (_Float16)f.y;
        bf[i + 2] = (_Float16)f.z; bf[i + 3] = (_Float16)f.w;
      }
      acc = wmma16(aq[kc], bf, acc);
    }
#pragma unroll
    for (int r = 0; r < 8; ++r) {
      int row = wave * 16 + ch * 8 + r;
      int qg  = q0 + row;
      float v = acc[r] * SCALE_;
      if (qg < key) v = -1.0e9f;   // causal mask (start_pos==0)
      sc[row * 512 + key] = v;
    }
  }

  // ---- softmax + gated prompt scores (one row per lane 0..15) ----
  if (lane < 16) {
    int row = wave * 16 + lane;
    float* srow = sc + row * 512;
    float mx = -1.0e30f;
    for (int k = 0; k < S_; ++k) mx = fmaxf(mx, srow[k]);
    float sum = 0.0f;
    for (int k = 0; k < S_; ++k) { float e = __expf(srow[k] - mx); srow[k] = e; sum += e; }
    float inv = 1.0f / sum;
    for (int k = 0; k < S_; ++k) pr[row * 512 + k] = (_Float16)(srow[k] * inv);

    float pscr[P_];
    float pmx = -1.0e30f;
    for (int p = 0; p < P_; ++p) {
      const float* pk = PK + (size_t)(b * P_ + p) * D_ + h * HD_;
      float dot = 0.0f;
      for (int d = 0; d < HD_; ++d) dot += (float)qs[row * 128 + d] * pk[d];
      pscr[p] = dot * SCALE_;
      pmx = fmaxf(pmx, pscr[p]);
    }
    float psum = 0.0f;
    for (int p = 0; p < P_; ++p) { pscr[p] = __expf(pscr[p] - pmx); psum += pscr[p]; }
    float g = tanhf(gate[h]) * new_gate[0] / psum;
    for (int p = 0; p < P_; ++p) ps[row * P_ + p] = pscr[p] * g;
  }
  __syncthreads();

  // ---- stage V transposed [d][key] as f16 (reuses score region) ----
  for (int k = tid; k < S_; k += 128) {
    const float* vrow = Vc + (size_t)(b * S_ + k) * D_ + h * HD_;
    for (int d = 0; d < HD_; d += 4) {
      float4 f = *(const float4*)(vrow + d);
      vt[(d + 0) * 512 + k] = (_Float16)f.x;
      vt[(d + 1) * 512 + k] = (_Float16)f.y;
      vt[(d + 2) * 512 + k] = (_Float16)f.z;
      vt[(d + 3) * 512 + k] = (_Float16)f.w;
    }
  }
  __syncthreads();

  // ---- out = P @ V (16x512 @ 512x128) + prompt probs @ PV ----
  v8f oacc[8];
#pragma unroll
  for (int nt = 0; nt < 8; ++nt) oacc[nt] = wmma_zero();
  for (int k0 = 0; k0 < S_; k0 += 32) {
    v16h af = lds_a_frag(pr + (wave * 16) * 512, 512, k0);
#pragma unroll
    for (int nt = 0; nt < 8; ++nt) {
      v16h bf = lds_b_frag(vt + (nt * 16) * 512, 512, k0);
      oacc[nt] = wmma16(af, bf, oacc[nt]);
    }
  }

#pragma unroll
  for (int nt = 0; nt < 8; ++nt) {
    int d = nt * 16 + cl;
    float pvcol[P_];
#pragma unroll
    for (int p = 0; p < P_; ++p)
      pvcol[p] = PV[(size_t)(b * P_ + p) * D_ + h * HD_ + d];
#pragma unroll
    for (int r = 0; r < 8; ++r) {
      int row = wave * 16 + ch * 8 + r;
      float v = oacc[nt][r];
      const float* psr = ps + row * P_;
#pragma unroll
      for (int p = 0; p < P_; ++p) v += psr[p] * pvcol[p];
      O[(size_t)(b * S_ + q0 + row) * D_ + h * HD_ + d] = v;
    }
  }
}

// ---------------------------------------------------------------------------
// Small bandwidth-bound kernels
// ---------------------------------------------------------------------------
__global__ __launch_bounds__(256)
void rmsnorm_k(const float* __restrict__ x, const float* __restrict__ w,
               float* __restrict__ out) {
  __shared__ float red[8];
  __shared__ float scale_s;
  int row = blockIdx.x;
  const float* xr = x + (size_t)row * D_;
  float s = 0.0f;
  for (int i = threadIdx.x; i < D_; i += 256) { float v = xr[i]; s += v * v; }
#pragma unroll
  for (int off = 16; off > 0; off >>= 1) s += __shfl_down(s, off, 32);
  int lane = threadIdx.x & 31, wave = threadIdx.x >> 5;
  if (lane == 0) red[wave] = s;
  __syncthreads();
  if (threadIdx.x == 0) {
    float t = 0.0f;
    for (int i = 0; i < 8; ++i) t += red[i];
    scale_s = rsqrtf(t / (float)D_ + EPS_);
  }
  __syncthreads();
  float sc = scale_s;
  for (int i = threadIdx.x; i < D_; i += 256)
    out[(size_t)row * D_ + i] = xr[i] * sc * w[i];
}

__global__ void rope_k(float* __restrict__ t, const float* __restrict__ cs,
                       const float* __restrict__ sn) {
  int idx = blockIdx.x * blockDim.x + threadIdx.x;
  if (idx >= B_ * S_ * H_ * (HD_ / 2)) return;
  int i = idx & 63;
  int h = (idx >> 6) & 31;
  int s = (idx >> 11) & 511;
  int b = idx >> 20;
  size_t base = (size_t)(b * S_ + s) * D_ + h * HD_ + 2 * i;
  float a = t[base], bb = t[base + 1];
  float c = cs[s * 64 + i], si = sn[s * 64 + i];
  t[base]     = a * c - bb * si;
  t[base + 1] = a * si + bb * c;
}

__global__ void lora_down_k(const float* __restrict__ A, const float* __restrict__ L1,
                            float* __restrict__ T, int M, int K) {
  int idx = blockIdx.x * blockDim.x + threadIdx.x;
  if (idx >= M * R_) return;
  int m = idx >> 4, r = idx & 15;
  const float* a = A + (size_t)m * K;
  const float* l = L1 + (size_t)r * K;
  float s = 0.0f;
  for (int k = 0; k < K; k += 4) {
    float4 av = *(const float4*)(a + k);
    float4 lv = *(const float4*)(l + k);
    s += av.x * lv.x + av.y * lv.y + av.z * lv.z + av.w * lv.w;
  }
  T[idx] = s;
}

__global__ void lora_up_k(const float* __restrict__ T, const float* __restrict__ L2,
                          float* __restrict__ C, int M, int N) {
  long long idx = (long long)blockIdx.x * blockDim.x + threadIdx.x;
  if (idx >= (long long)M * N) return;
  int m = (int)(idx / N), n = (int)(idx % N);
  const float* t = T + m * R_;
  const float* l = L2 + (size_t)n * R_;
  float s = 0.0f;
#pragma unroll
  for (int r = 0; r < R_; ++r) s += t[r] * l[r];
  C[idx] += s;
}

__global__ void add_k(const float* __restrict__ a, const float* __restrict__ b,
                      float* __restrict__ o, long long n) {
  long long i = (long long)blockIdx.x * blockDim.x + threadIdx.x;
  if (i < n) o[i] = a[i] + b[i];
}

__global__ void silu_mul_k(float* __restrict__ a1, const float* __restrict__ a3,
                           long long n) {
  long long i = (long long)blockIdx.x * blockDim.x + threadIdx.x;
  if (i < n) {
    float v = a1[i];
    a1[i] = (v / (1.0f + __expf(-v))) * a3[i];
  }
}

// ---------------------------------------------------------------------------
extern "C" void kernel_launch(void* const* d_in, const int* in_sizes, int n_in,
                              void* d_out, int out_size, void* d_ws, size_t ws_size,
                              hipStream_t stream) {
  (void)in_sizes; (void)n_in; (void)out_size; (void)ws_size;
  const float* x           = (const float*)d_in[0];
  const float* prompt      = (const float*)d_in[1];
  const float* freqs_cos   = (const float*)d_in[5];
  const float* freqs_sin   = (const float*)d_in[6];
  const float* wq_w = (const float*)d_in[7];
  const float* wq_b = (const float*)d_in[8];
  const float* wk_w = (const float*)d_in[9];
  const float* wv_w = (const float*)d_in[10];
  const float* wo_w = (const float*)d_in[11];
  const float* wo_b = (const float*)d_in[12];
  const float* lq1  = (const float*)d_in[13];
  const float* lq2  = (const float*)d_in[14];
  const float* lk1  = (const float*)d_in[15];
  const float* lk2  = (const float*)d_in[16];
  const float* lv1  = (const float*)d_in[17];
  const float* lv2  = (const float*)d_in[18];
  const float* lo1  = (const float*)d_in[19];
  const float* lo2  = (const float*)d_in[20];
  const float* gate      = (const float*)d_in[21];
  const float* new_gate  = (const float*)d_in[22];
  const float* attn_norm = (const float*)d_in[23];
  const float* ffn_norm  = (const float*)d_in[24];
  const float* w1_w = (const float*)d_in[25];
  const float* w1_b = (const float*)d_in[26];
  const float* w2_w = (const float*)d_in[27];
  const float* w2_b = (const float*)d_in[28];
  const float* w3_w = (const float*)d_in[29];
  const float* w3_b = (const float*)d_in[30];
  const float* fl1a = (const float*)d_in[31];
  const float* fl1b = (const float*)d_in[32];
  const float* fl2a = (const float*)d_in[33];
  const float* fl2b = (const float*)d_in[34];
  const float* fl3a = (const float*)d_in[35];
  const float* fl3b = (const float*)d_in[36];

  const int M = B_ * S_;  // 2048
  char* wp = (char*)d_ws;
  size_t MB = (size_t)M * D_ * sizeof(float);    // 32 MiB
  size_t HB = (size_t)M * HID_ * sizeof(float);  // 86 MiB
  float* buf_h  = (float*)wp; wp += MB;  // h / attn-raw-out / ffn-out
  float* buf_q  = (float*)wp; wp += MB;  // xq / attn-proj
  float* buf_k  = (float*)wp; wp += MB;  // xk / h2
  float* buf_v  = (float*)wp; wp += MB;  // xv / g
  float* buf_a1 = (float*)wp; wp += HB;
  float* buf_a3 = (float*)wp; wp += HB;
  float* buf_t  = (float*)wp; wp += (size_t)M * R_ * sizeof(float);
  float* buf_pk = (float*)wp; wp += (size_t)B_ * P_ * D_ * sizeof(float);
  float* buf_pv = (float*)wp; wp += (size_t)B_ * P_ * D_ * sizeof(float);

  auto gemm = [&](const float* A, const float* W_, const float* bias, float* C,
                  int m, int n, int k) {
    dim3 g(n / 128, (m + 127) / 128);
    gemm_f16<<<g, 256, 131072, stream>>>(A, W_, bias, C, m, n, k);
  };
  auto lora = [&](const float* A, const float* l1, const float* l2, float* C,
                  int m, int n, int k) {
    int cnt = m * R_;
    lora_down_k<<<(cnt + 255) / 256, 256, 0, stream>>>(A, l1, buf_t, m, k);
    long long c2 = (long long)m * n;
    lora_up_k<<<(int)((c2 + 255) / 256), 256, 0, stream>>>(buf_t, l2, C, m, n);
  };

  // attention branch
  rmsnorm_k<<<M, 256, 0, stream>>>(x, attn_norm, buf_h);
  gemm(buf_h, wq_w, wq_b, buf_q, M, D_, D_);
  gemm(buf_h, wk_w, nullptr, buf_k, M, D_, D_);
  gemm(buf_h, wv_w, nullptr, buf_v, M, D_, D_);
  lora(buf_h, lq1, lq2, buf_q, M, D_, D_);
  lora(buf_h, lk1, lk2, buf_k, M, D_, D_);
  lora(buf_h, lv1, lv2, buf_v, M, D_, D_);
  int rn = B_ * S_ * H_ * (HD_ / 2);
  rope_k<<<(rn + 255) / 256, 256, 0, stream>>>(buf_q, freqs_cos, freqs_sin);
  rope_k<<<(rn + 255) / 256, 256, 0, stream>>>(buf_k, freqs_cos, freqs_sin);
  gemm(prompt, wk_w, nullptr, buf_pk, B_ * P_, D_, D_);
  gemm(prompt, wv_w, nullptr, buf_pv, B_ * P_, D_, D_);
  {
    dim3 g(S_ / 64, H_, B_);
    size_t smem = 64 * 128 * 2       // q f16
                + 64 * 512 * 4       // scores f32 / Vt f16
                + 64 * 512 * 2       // probs f16
                + 64 * P_ * 4;       // prompt probs
    attn_f16<<<g, 128, smem, stream>>>(buf_q, buf_k, buf_v, buf_pk, buf_pv,
                                       gate, new_gate, buf_h);
  }
  gemm(buf_h, wo_w, wo_b, buf_q, M, D_, D_);
  lora(buf_h, lo1, lo2, buf_q, M, D_, D_);
  long long nd = (long long)M * D_;
  add_k<<<(int)((nd + 255) / 256), 256, 0, stream>>>(x, buf_q, buf_k, nd);  // h2

  // FFN branch
  rmsnorm_k<<<M, 256, 0, stream>>>(buf_k, ffn_norm, buf_v);  // g
  gemm(buf_v, w1_w, w1_b, buf_a1, M, HID_, D_);
  lora(buf_v, fl1a, fl1b, buf_a1, M, HID_, D_);
  gemm(buf_v, w3_w, w3_b, buf_a3, M, HID_, D_);
  lora(buf_v, fl3a, fl3b, buf_a3, M, HID_, D_);
  long long nh = (long long)M * HID_;
  silu_mul_k<<<(int)((nh + 255) / 256), 256, 0, stream>>>(buf_a1, buf_a3, nh);
  gemm(buf_a1, w2_w, w2_b, buf_h, M, D_, HID_);
  lora(buf_a1, fl2a, fl2b, buf_h, M, D_, HID_);

  add_k<<<(int)((nd + 255) / 256), 256, 0, stream>>>(buf_k, buf_h, (float*)d_out, nd);
}
